// MultiHeadAttention_76046690943475
// MI455X (gfx1250) — matrix operand order repaired and checked by reading, not compiled, not run on hardware
//
#include <hip/hip_runtime.h>
#include <hip/hip_bf16.h>

#define EMBED 1024
#define HEADS 16
#define HDIM  64
#define BATCH 2
#define SEQ   2048
#define MROWS (BATCH * SEQ)       /* 4096 */
#define ATTN_SCALE 0.125f         /* 1/sqrt(64) */

typedef __attribute__((ext_vector_type(16))) __bf16 v16bf;
typedef __attribute__((ext_vector_type(8)))  __bf16 v8bf;
typedef __attribute__((ext_vector_type(8)))  float  v8f;

// ---------------------------------------------------------------------------
// WMMA fragment loaders (wave32; A/B share one contiguous k order, not the ISA table order)
// A (16x32 bf16, MxK): lane half h=l>>4: elem i(0..15) -> K=16h+i (contiguous)
// B (32x16 bf16, KxN): lane half h:      elem i(0..15) -> K=16h+i (contiguous)
// C/D (16x16 f32):     lane l: col n = l&15, vgpr j -> row m = 8*(l>>4)+j
// ---------------------------------------------------------------------------
__device__ __forceinline__ v16bf load_frag_a(const __bf16* __restrict__ rowk, int h8) {
  v8bf lo = *(const v8bf*)(rowk + 2*h8);
  v8bf hi = *(const v8bf*)(rowk + 2*h8 + 8);
  return __builtin_shufflevector(lo, hi, 0,1,2,3,4,5,6,7,8,9,10,11,12,13,14,15);
}
__device__ __forceinline__ v16bf load_frag_b(const __bf16* __restrict__ rowk, int h16) {
  v8bf lo = *(const v8bf*)(rowk + h16);
  v8bf hi = *(const v8bf*)(rowk + h16 + 8);
  return __builtin_shufflevector(lo, hi, 0,1,2,3,4,5,6,7,8,9,10,11,12,13,14,15);
}
__device__ __forceinline__ v8f wmma_bf(v16bf a, v16bf b, v8f c) {
  return __builtin_amdgcn_wmma_f32_16x16x32_bf16(false, a, false, b, (short)0, c,
                                                 false, false);
}
__device__ __forceinline__ v8f vzero8() {
  v8f z = {0.f,0.f,0.f,0.f,0.f,0.f,0.f,0.f};
  return z;
}

// ---------------------------------------------------------------------------
// Kernel 0: fp32 -> bf16 convert, 8 elements/thread (2x b128 load, 1x b128 store)
// ---------------------------------------------------------------------------
__global__ __launch_bounds__(256)
void mha_cvt_bf16(const float* __restrict__ in, __bf16* __restrict__ out, int n8) {
  const int i = blockIdx.x * 256 + threadIdx.x;
  if (i < n8) {
    v8f x = ((const v8f*)in)[i];
    v8bf y;
    #pragma unroll
    for (int j = 0; j < 8; ++j) y[j] = (__bf16)x[j];
    ((v8bf*)out)[i] = y;
  }
}

// ---------------------------------------------------------------------------
// Kernel 1: fused QKV projection.  Y = Xb @ W^T + bias  (W row-major [N,K])
// grid = (E/128, M/128, 3), block = 256 (8 waves). Each wave: 2x4 16x16 tiles.
// which==0 -> Qb [B,H,S,64]; which==1 -> Kb [B,H,S,64]; which==2 -> Vt [B,H,64,S]
// ---------------------------------------------------------------------------
__global__ __launch_bounds__(256)
void mha_qkv_proj(const __bf16* __restrict__ Xb, const __bf16* __restrict__ W3b,
                  const float* __restrict__ bq, const float* __restrict__ bk,
                  const float* __restrict__ bv,
                  __bf16* __restrict__ Qb, __bf16* __restrict__ Kb,
                  __bf16* __restrict__ Vt)
{
  const int which = blockIdx.z;
  const __bf16* __restrict__ W = W3b + (size_t)which * EMBED * EMBED;
  const int n0 = blockIdx.x * 128;
  const int m0 = blockIdx.y * 128;
  const int wave = threadIdx.x >> 5;
  const int lane = threadIdx.x & 31;
  const int wm = wave >> 1;              // 0..3 : 32 rows each
  const int wn = wave & 1;               // 0..1 : 64 cols each
  const int r   = lane & 15;
  const int h8  = (lane >> 4) * 8;
  const int h16 = (lane >> 4) * 16;

  v8f acc[2][4];
  #pragma unroll
  for (int mt = 0; mt < 2; ++mt)
    #pragma unroll
    for (int nt = 0; nt < 4; ++nt) acc[mt][nt] = vzero8();

  const __bf16* __restrict__ arow0 = Xb + (size_t)(m0 + wm*32 +  0 + r) * EMBED;
  const __bf16* __restrict__ arow1 = Xb + (size_t)(m0 + wm*32 + 16 + r) * EMBED;
  const __bf16* brow[4];
  #pragma unroll
  for (int nt = 0; nt < 4; ++nt)
    brow[nt] = W + (size_t)(n0 + wn*64 + nt*16 + r) * EMBED;

  for (int k = 0; k < EMBED; k += 32) {
    // speculative prefetch one cacheline ahead on the streaming activations
    __builtin_prefetch(arow0 + k + 64, 0, 1);
    __builtin_prefetch(arow1 + k + 64, 0, 1);
    v16bf af[2], bf[4];
    af[0] = load_frag_a(arow0 + k, h8);
    af[1] = load_frag_a(arow1 + k, h8);
    #pragma unroll
    for (int nt = 0; nt < 4; ++nt) bf[nt] = load_frag_b(brow[nt] + k, h16);
    #pragma unroll
    for (int mt = 0; mt < 2; ++mt)
      #pragma unroll
      for (int nt = 0; nt < 4; ++nt)
        acc[mt][nt] = wmma_bf(af[mt], bf[nt], acc[mt][nt]);
  }

  // Branch-free epilogue: all three layouts are
  //   base[(b*H+h)*S*Dh + d*sD + s*sS]
  const float* __restrict__ bias = (which == 0) ? bq : (which == 1) ? bk : bv;
  __bf16* __restrict__ base = (which == 0) ? Qb : (which == 1) ? Kb : Vt;
  const int sD = (which == 2) ? SEQ : 1;
  const int sS = (which == 2) ? 1 : HDIM;

  #pragma unroll
  for (int nt = 0; nt < 4; ++nt) {
    const int ncol = n0 + wn*64 + nt*16 + r;
    const float bb = bias[ncol];
    const int hh = ncol >> 6;              // head
    const int d  = ncol & (HDIM - 1);      // dim in head
    #pragma unroll
    for (int mt = 0; mt < 2; ++mt) {
      #pragma unroll
      for (int j = 0; j < 8; ++j) {
        const int mrow = m0 + wm*32 + mt*16 + h8 + j;
        const int b = mrow >> 11;          // SEQ = 2048
        const int s = mrow & (SEQ - 1);
        base[(size_t)(b*HEADS + hh) * (SEQ * HDIM) + (size_t)d * sD + (size_t)s * sS]
            = (__bf16)(acc[mt][nt][j] + bb);
      }
    }
  }
}

// ---------------------------------------------------------------------------
// Kernel 2: flash attention, register-resident, transposed-score trick.
// grid = (S/64, B*H), block = 128 (4 independent waves, 16 queries each).
//   S^T tile = wmma(A=K rows, B=Q rows)    -> queries live in lane columns
//   O^T acc  = wmma(A=V^T rows, B=P^T)     -> softmax rescale is per-lane scalar
// ---------------------------------------------------------------------------
__global__ __launch_bounds__(128)
void mha_flash_attn(const __bf16* __restrict__ Qb, const __bf16* __restrict__ Kb,
                    const __bf16* __restrict__ Vt, __bf16* __restrict__ A2b,
                    const int* __restrict__ is_causal)
{
  const int bh = blockIdx.y;
  const int b = bh / HEADS;
  const int h = bh % HEADS;
  const int wave = threadIdx.x >> 5;
  const int lane = threadIdx.x & 31;
  const int q0 = blockIdx.x * 64 + wave * 16;
  const int r = lane & 15;
  const int h8 = (lane >> 4) * 8;
  const int h16 = (lane >> 4) * 16;
  const bool hiHalf = (lane >= 16);

  const __bf16* __restrict__ Q = Qb + (size_t)bh * SEQ * HDIM;
  const __bf16* __restrict__ K = Kb + (size_t)bh * SEQ * HDIM;
  const __bf16* __restrict__ V = Vt + (size_t)bh * HDIM * SEQ;   // [64][S]

  // Q as B-fragments over the two Dh halves (dk*32): held for whole kernel
  v16bf qf[2];
  #pragma unroll
  for (int dk = 0; dk < 2; ++dk)
    qf[dk] = load_frag_b(Q + (size_t)(q0 + r) * HDIM + dk*32, h16);

  const bool causal = (is_causal[0] != 0);
  const int kend = causal ? (q0 + 16) : SEQ;   // wave-uniform trip count
  const int qcol = q0 + r;                     // query handled by this lane column

  float m_i = -__builtin_inff();
  float l_i = 0.f;
  v8f ot[4];
  #pragma unroll
  for (int c = 0; c < 4; ++c) ot[c] = vzero8();

  for (int kb = 0; kb < kend; kb += 32) {
    // scores^T: keys kb..kb+15 (st0), kb+16..kb+31 (st1)
    v8f st0 = vzero8(), st1 = vzero8();
    #pragma unroll
    for (int dk = 0; dk < 2; ++dk) {
      v16bf kf0 = load_frag_a(K + (size_t)(kb +      r) * HDIM + dk*32, h8);
      v16bf kf1 = load_frag_a(K + (size_t)(kb + 16 + r) * HDIM + dk*32, h8);
      st0 = wmma_bf(kf0, qf[dk], st0);
      st1 = wmma_bf(kf1, qf[dk], st1);
    }
    // scale + causal mask; row (=query) stats via 8 regs + one shfl_xor(16)
    float sv[16];
    float vmax = -__builtin_inff();
    #pragma unroll
    for (int j = 0; j < 8; ++j) {
      float x0 = st0[j] * ATTN_SCALE;
      float x1 = st1[j] * ATTN_SCALE;
      const int k0i = kb +      h8 + j;
      const int k1i = kb + 16 + h8 + j;
      x0 = (causal && (k0i > qcol)) ? -__builtin_inff() : x0;
      x1 = (causal && (k1i > qcol)) ? -__builtin_inff() : x1;
      sv[j] = x0; sv[8 + j] = x1;
      vmax = fmaxf(vmax, fmaxf(x0, x1));
    }
    vmax = fmaxf(vmax, __shfl_xor(vmax, 16, 32));
    const float m_new = fmaxf(m_i, vmax);
    const float alpha = __expf(m_i - m_new);   // 0 on first chunk (m_i = -inf)

    float p0[8], p1[8];
    float psum = 0.f;
    #pragma unroll
    for (int j = 0; j < 8; ++j) {
      p0[j] = __expf(sv[j]     - m_new);
      p1[j] = __expf(sv[8 + j] - m_new);
      psum += p0[j] + p1[j];
    }
    psum += __shfl_xor(psum, 16, 32);
    l_i = l_i * alpha + psum;
    m_i = m_new;

    // Repack P^T into a B-fragment: B elem i -> key kb+16h+i.
    // S^T acc holds (query l&15, key kb + t*16 + 8h + j): exchange the
    // half-tiles across the (l, l+16) lane pair -> 8 shuffles.
    v16bf pf;
    #pragma unroll
    for (int j = 0; j < 8; ++j) {
      const float send = hiHalf ? p0[j] : p1[j];
      const float recv = __shfl_xor(send, 16, 32);
      const float lo = hiHalf ? recv  : p0[j];
      const float hi = hiHalf ? p1[j] : recv;
      pf[j]     = (__bf16)lo;
      pf[8 + j] = (__bf16)hi;
    }

    // O^T += V^T(16xDh-tile x 32keys) @ P^T ; rescale is per-lane uniform
    #pragma unroll
    for (int c = 0; c < 4; ++c) {
      v16bf vf = load_frag_a(V + (size_t)(c*16 + r) * SEQ + kb, h8);
      #pragma unroll
      for (int j = 0; j < 8; ++j) ot[c][j] *= alpha;
      ot[c] = wmma_bf(vf, pf, ot[c]);
    }
  }

  // normalize + merge heads into attn-out [B,S,E] bf16
  const float inv = 1.f / l_i;     // identical across lane pair
  __bf16* __restrict__ dst = A2b + (size_t)(b * SEQ + q0 + r) * EMBED + h * HDIM;
  #pragma unroll
  for (int c = 0; c < 4; ++c)
    #pragma unroll
    for (int j = 0; j < 8; ++j)
      dst[c*16 + h8 + j] = (__bf16)(ot[c][j] * inv);
}

// ---------------------------------------------------------------------------
// Kernel 3: output projection.  out = Ab @ Wo^T + bo  (fp32 result)
// ---------------------------------------------------------------------------
__global__ __launch_bounds__(256)
void mha_out_proj(const __bf16* __restrict__ Ab, const __bf16* __restrict__ Wob,
                  const float* __restrict__ bo, float* __restrict__ out)
{
  const int n0 = blockIdx.x * 128;
  const int m0 = blockIdx.y * 128;
  const int wave = threadIdx.x >> 5;
  const int lane = threadIdx.x & 31;
  const int wm = wave >> 1;
  const int wn = wave & 1;
  const int r   = lane & 15;
  const int h8  = (lane >> 4) * 8;
  const int h16 = (lane >> 4) * 16;

  v8f acc[2][4];
  #pragma unroll
  for (int mt = 0; mt < 2; ++mt)
    #pragma unroll
    for (int nt = 0; nt < 4; ++nt) acc[mt][nt] = vzero8();

  const __bf16* __restrict__ arow0 = Ab + (size_t)(m0 + wm*32 +  0 + r) * EMBED;
  const __bf16* __restrict__ arow1 = Ab + (size_t)(m0 + wm*32 + 16 + r) * EMBED;
  const __bf16* brow[4];
  #pragma unroll
  for (int nt = 0; nt < 4; ++nt)
    brow[nt] = Wob + (size_t)(n0 + wn*64 + nt*16 + r) * EMBED;

  for (int k = 0; k < EMBED; k += 32) {
    __builtin_prefetch(arow0 + k + 64, 0, 1);
    __builtin_prefetch(arow1 + k + 64, 0, 1);
    v16bf af[2], bf[4];
    af[0] = load_frag_a(arow0 + k, h8);
    af[1] = load_frag_a(arow1 + k, h8);
    #pragma unroll
    for (int nt = 0; nt < 4; ++nt) bf[nt] = load_frag_b(brow[nt] + k, h16);
    #pragma unroll
    for (int mt = 0; mt < 2; ++mt)
      #pragma unroll
      for (int nt = 0; nt < 4; ++nt)
        acc[mt][nt] = wmma_bf(af[mt], bf[nt], acc[mt][nt]);
  }

  #pragma unroll
  for (int nt = 0; nt < 4; ++nt) {
    const int ncol = n0 + wn*64 + nt*16 + r;
    const float bb = bo[ncol];
    #pragma unroll
    for (int mt = 0; mt < 2; ++mt) {
      #pragma unroll
      for (int j = 0; j < 8; ++j) {
        const int mrow = m0 + wm*32 + mt*16 + h8 + j;
        out[(size_t)mrow * EMBED + ncol] = acc[mt][nt][j] + bb;
      }
    }
  }
}

// ---------------------------------------------------------------------------
extern "C" void kernel_launch(void* const* d_in, const int* in_sizes, int n_in,
                              void* d_out, int out_size, void* d_ws, size_t ws_size,
                              hipStream_t stream)
{
  (void)in_sizes; (void)n_in; (void)out_size; (void)ws_size;

  const float* QKV = (const float*)d_in[0];
  const float* Wq  = (const float*)d_in[1];
  const float* bq  = (const float*)d_in[2];
  const float* Wk  = (const float*)d_in[3];
  const float* bk  = (const float*)d_in[4];
  const float* Wv  = (const float*)d_in[5];
  const float* bv  = (const float*)d_in[6];
  const float* Wo  = (const float*)d_in[7];
  const float* bo  = (const float*)d_in[8];
  const int* causal = (const int*)d_in[9];

  char* p = (char*)d_ws;
  __bf16* Xb  = (__bf16*)p; p += (size_t)MROWS * EMBED * 2;        //  8 MB
  __bf16* W3b = (__bf16*)p; p += (size_t)3 * EMBED * EMBED * 2;    //  6 MB
  __bf16* Wob = (__bf16*)p; p += (size_t)EMBED * EMBED * 2;        //  2 MB
  __bf16* Qb  = (__bf16*)p; p += (size_t)MROWS * EMBED * 2;        //  8 MB
  __bf16* Kb  = (__bf16*)p; p += (size_t)MROWS * EMBED * 2;        //  8 MB
  __bf16* Vt  = (__bf16*)p; p += (size_t)MROWS * EMBED * 2;        //  8 MB
  __bf16* A2b = (__bf16*)p; p += (size_t)MROWS * EMBED * 2;        //  8 MB

  const int nX8 = (MROWS * EMBED) / 8;
  const int nW8 = (EMBED * EMBED) / 8;
  const int nW  = EMBED * EMBED;
  mha_cvt_bf16<<<(nX8 + 255) / 256, 256, 0, stream>>>(QKV, Xb, nX8);
  mha_cvt_bf16<<<(nW8 + 255) / 256, 256, 0, stream>>>(Wq, W3b,        nW8);
  mha_cvt_bf16<<<(nW8 + 255) / 256, 256, 0, stream>>>(Wk, W3b + nW,   nW8);
  mha_cvt_bf16<<<(nW8 + 255) / 256, 256, 0, stream>>>(Wv, W3b + 2*nW, nW8);
  mha_cvt_bf16<<<(nW8 + 255) / 256, 256, 0, stream>>>(Wo, Wob,        nW8);

  dim3 gp(EMBED / 128, MROWS / 128, 3);
  mha_qkv_proj<<<gp, 256, 0, stream>>>(Xb, W3b, bq, bk, bv, Qb, Kb, Vt);

  dim3 ga(SEQ / 64, BATCH * HEADS);
  mha_flash_attn<<<ga, 128, 0, stream>>>(Qb, Kb, Vt, A2b, causal);

  dim3 go(EMBED / 128, MROWS / 128);
  mha_out_proj<<<go, 256, 0, stream>>>(A2b, Wob, bo, (float*)d_out);
}
